// GraphTransformerLayer_80668075753491
// MI455X (gfx1250) — matrix operand order repaired and checked by reading, not compiled, not run on hardware
//
#include <hip/hip_runtime.h>

// ---------------------------------------------------------------------------
// GraphTransformer layer for gfx1250 (MI455X), wave32 + WMMA f16.
// N=50000 nodes, E=640000 edges, C=128, H=4, D=32, ED=64, R=50.
// All dense projections via v_wmma_f32_16x16x32_f16; edge lin_edge folded
// into a 51-row rel_proj table (rel_emb @ We precomputed on device).
// ---------------------------------------------------------------------------

typedef _Float16 h16;
typedef __attribute__((ext_vector_type(16))) _Float16 v16h;
typedef __attribute__((ext_vector_type(8)))  float    v8f;

#define NNODES 50000
#define NEDGES 640000
#define NREL   51     /* R+1 rows in rel_emb */

union AV { v16h v; uint32_t u[8]; };

__device__ __forceinline__ float wave_sum32(float x) {
  #pragma unroll
  for (int o = 16; o > 0; o >>= 1) x += __shfl_xor(x, o, 32);
  return x;
}

__device__ __forceinline__ void atomicMaxF(float* addr, float val) {
  if (val >= 0.0f) atomicMax(reinterpret_cast<int*>(addr), __float_as_int(val));
  else             atomicMin(reinterpret_cast<unsigned int*>(addr), __float_as_uint(val));
}

// One 16x16 output tile, A row-major [M][lda] f16, B packed column-major
// Bt[col][ldb] f16 (so a (K,K+1) pair for one column is one aligned b32 load).
__device__ __forceinline__ v8f wmma_tile(const h16* __restrict__ A, int lda,
                                         const h16* __restrict__ Bt, int ldb,
                                         int row0, int col0, int K, int lane) {
  const int half = lane >> 4, mn = lane & 15;
  v8f c = {};
  const h16* arow = A + (size_t)(row0 + mn) * lda;
  const h16* bcol = Bt + (size_t)(col0 + mn) * ldb;
  for (int kc = 0; kc < K; kc += 32) {
    AV a, b;
    #pragma unroll
    for (int vv = 0; vv < 8; ++vv) {           // A 16x32 f16 layout (ISA 7.12.2)
      int grp = vv >> 2, j = vv & 3;
      int ka = kc + grp * 16 + half * 8 + 2 * j;
      a.u[vv] = *reinterpret_cast<const uint32_t*>(arow + ka);
    }
    #pragma unroll
    for (int vv = 0; vv < 8; ++vv) {           // B 32x16 f16: same K order as A
      int kb = kc + (vv >> 2) * 16 + half * 8 + 2 * (vv & 3);
      b.u[vv] = *reinterpret_cast<const uint32_t*>(bcol + kb);
    }
    c = __builtin_amdgcn_wmma_f32_16x16x32_f16(false, a.v, false, b.v,
                                               (short)0, c, false, false);
  }
  return c;
}

// --- prep: f32 -> f16 packed-transposed weights -----------------------------
__global__ void pack_weights_k(const float* __restrict__ Wq, const float* __restrict__ Wk,
                               const float* __restrict__ Wv, const float* __restrict__ Ws,
                               const float* __restrict__ W1, const float* __restrict__ W2,
                               h16* __restrict__ Wqkvs_t, h16* __restrict__ W1t,
                               h16* __restrict__ W2t) {
  int i = blockIdx.x * blockDim.x + threadIdx.x;
  if (i >= 512 * 128) return;
  // Wqkvs_t: [col 0..511][k 0..127]; cols = Wq|Wk|Wv|Ws
  int col = i >> 7, kk = i & 127;
  int sel = col >> 7, cc = col & 127;
  const float* W = sel == 0 ? Wq : sel == 1 ? Wk : sel == 2 ? Wv : Ws;
  Wqkvs_t[i] = (h16)W[kk * 128 + cc];
  // W1t: [col 0..511][k 0..127]; W1 is [128][512]
  W1t[i] = (h16)W1[kk * 512 + col];
  // W2t: [col 0..127][k 0..511]; W2 is [512][128]
  int col2 = i >> 9, k2 = i & 511;
  W2t[(size_t)col2 * 512 + k2] = (h16)W2[k2 * 128 + col2];
}

// rel_proj[r][c] = sum_k rel_emb[r][k] * We[k][c]   (51 x 64 @ 64 x 128)
__global__ void relproj_k(const float* __restrict__ rel_emb,
                          const float* __restrict__ We, float* __restrict__ relp) {
  int c = threadIdx.x;  // 128 threads
  for (int r = 0; r < NREL; ++r) {
    float a = 0.0f;
    for (int kk = 0; kk < 64; ++kk) a += rel_emb[r * 64 + kk] * We[kk * 128 + c];
    relp[r * 128 + c] = a;
  }
}

// --- norm1 -> f16 activations (one wave per row) ----------------------------
__global__ void ln1_k(const float* __restrict__ x, const float* __restrict__ g,
                      const float* __restrict__ b, h16* __restrict__ xnh, int n) {
  int wv = threadIdx.x >> 5, lane = threadIdx.x & 31;
  int row = blockIdx.x * 8 + wv;
  if (row >= n) return;
  const float* xr = x + (size_t)row * 128;
  float vals[4], s = 0.0f, s2 = 0.0f;
  #pragma unroll
  for (int i = 0; i < 4; ++i) {
    float t = xr[lane + 32 * i];
    vals[i] = t; s += t; s2 += t * t;
  }
  s = wave_sum32(s); s2 = wave_sum32(s2);
  float mu = s * (1.0f / 128.0f);
  float var = s2 * (1.0f / 128.0f) - mu * mu;
  float rs = rsqrtf(var + 1e-5f);
  #pragma unroll
  for (int i = 0; i < 4; ++i) {
    int c = lane + 32 * i;
    xnh[(size_t)row * 128 + c] = (h16)((vals[i] - mu) * rs * g[c] + b[c]);
  }
}

// --- fused Q|K|V|skip GEMM: [N,128] x [128,512] -----------------------------
__global__ void gemm_qkvs_k(const h16* __restrict__ Xh, const h16* __restrict__ Wt,
                            const float* __restrict__ bq, const float* __restrict__ bk,
                            const float* __restrict__ bv, const float* __restrict__ bs,
                            float* __restrict__ q, float* __restrict__ k,
                            float* __restrict__ v, float* __restrict__ xr, int nrows) {
  int wv = threadIdx.x >> 5, lane = threadIdx.x & 31;
  int row0 = blockIdx.x * 16;
  if (row0 >= nrows) return;
  int half = lane >> 4, mn = lane & 15;
  for (int t = wv; t < 32; t += 8) {
    int col0 = t * 16;
    v8f c = wmma_tile(Xh, 128, Wt, 128, row0, col0, 128, lane);
    int col = col0 + mn;
    int sel = col >> 7, cc = col & 127;
    const float* bias = sel == 0 ? bq : sel == 1 ? bk : sel == 2 ? bv : bs;
    float* out = sel == 0 ? q : sel == 1 ? k : sel == 2 ? v : xr;
    float bb = bias[cc];
    #pragma unroll
    for (int r = 0; r < 8; ++r) {
      int m = row0 + half * 8 + r;
      out[(size_t)m * 128 + cc] = c[r] + bb;
    }
  }
}

// --- softmax state init ------------------------------------------------------
__global__ void init_k(float* __restrict__ nmax, float* __restrict__ nsum,
                       float* __restrict__ acc) {
  long i = (long)blockIdx.x * blockDim.x + threadIdx.x;
  if (i < (long)NNODES * 128) acc[i] = 0.0f;
  if (i < (long)NNODES * 4) { nmax[i] = -3.0e38f; nsum[i] = 0.0f; }
}

// --- alpha[e][h] = <q[dst], k[src]+e> / sqrt(D)  (one wave per edge) --------
__global__ void edge_alpha_k(const int* __restrict__ ei, const int* __restrict__ et,
                             const float* __restrict__ q, const float* __restrict__ k,
                             const float* __restrict__ relp, float* __restrict__ alpha) {
  int wv = threadIdx.x >> 5, lane = threadIdx.x & 31;
  long e = (long)blockIdx.x * 8 + wv;
  if (e >= NEDGES) return;
  int src = ei[e], dst = ei[NEDGES + e], r = et[e];
  const float* qd = q + (size_t)dst * 128;
  const float* ks = k + (size_t)src * 128;
  const float* ep = relp + (size_t)r * 128;
  float p[4];
  #pragma unroll
  for (int h = 0; h < 4; ++h) {
    int c = h * 32 + lane;
    p[h] = qd[c] * (ks[c] + ep[c]);
  }
  #pragma unroll
  for (int h = 0; h < 4; ++h) p[h] = wave_sum32(p[h]);
  if (lane == 0) {
    const float sc = 0.17677669529663687f;  // 1/sqrt(32)
    #pragma unroll
    for (int h = 0; h < 4; ++h) alpha[e * 4 + h] = p[h] * sc;
  }
}

__global__ void segmax_k(const int* __restrict__ ei, const float* __restrict__ alpha,
                         float* __restrict__ nmax) {
  long i = (long)blockIdx.x * blockDim.x + threadIdx.x;
  if (i >= (long)NEDGES * 4) return;
  long e = i >> 2; int h = (int)(i & 3);
  int dst = ei[NEDGES + e];
  atomicMaxF(nmax + (size_t)dst * 4 + h, alpha[i]);
}

__global__ void expsum_k(const int* __restrict__ ei, float* __restrict__ alpha,
                         const float* __restrict__ nmax, float* __restrict__ nsum) {
  long i = (long)blockIdx.x * blockDim.x + threadIdx.x;
  if (i >= (long)NEDGES * 4) return;
  long e = i >> 2; int h = (int)(i & 3);
  int dst = ei[NEDGES + e];
  float ea = __expf(alpha[i] - nmax[(size_t)dst * 4 + h]);
  alpha[i] = ea;                                   // overwrite with exp
  atomicAdd(nsum + (size_t)dst * 4 + h, ea);
}

// --- out[dst] += attn * (v[src] + e)  (one wave per edge, f32 atomics) ------
__global__ void agg_k(const int* __restrict__ ei, const int* __restrict__ et,
                      const float* __restrict__ alpha, const float* __restrict__ nsum,
                      const float* __restrict__ v, const float* __restrict__ relp,
                      float* __restrict__ acc) {
  int wv = threadIdx.x >> 5, lane = threadIdx.x & 31;
  long e = (long)blockIdx.x * 8 + wv;
  if (e >= NEDGES) return;
  int src = ei[e], dst = ei[NEDGES + e], r = et[e];
  const float* vs = v + (size_t)src * 128;
  const float* ep = relp + (size_t)r * 128;
  float* ad = acc + (size_t)dst * 128;
  #pragma unroll
  for (int h = 0; h < 4; ++h) {
    float attn = alpha[e * 4 + h] / (nsum[(size_t)dst * 4 + h] + 1e-16f);
    int c = h * 32 + lane;
    atomicAdd(ad + c, attn * (vs[c] + ep[c]));
  }
}

// --- beta gate + residual + norm2 (one wave per row) ------------------------
__global__ void gate_k(const float* __restrict__ x, const float* __restrict__ acc,
                       const float* __restrict__ xr, const float* __restrict__ wb,
                       const float* __restrict__ g2, const float* __restrict__ b2,
                       float* __restrict__ h, h16* __restrict__ h2h, int n) {
  int wv = threadIdx.x >> 5, lane = threadIdx.x & 31;
  int row = blockIdx.x * 8 + wv;
  if (row >= n) return;
  float o[4], rr[4], dot = 0.0f;
  #pragma unroll
  for (int i = 0; i < 4; ++i) {
    int c = lane + 32 * i;
    o[i]  = acc[(size_t)row * 128 + c];
    rr[i] = xr[(size_t)row * 128 + c];
    dot += o[i] * wb[c] + rr[i] * wb[128 + c] + (o[i] - rr[i]) * wb[256 + c];
  }
  dot = wave_sum32(dot);
  float beta = 1.0f / (1.0f + __expf(-dot));
  float hv[4], s = 0.0f, s2 = 0.0f;
  #pragma unroll
  for (int i = 0; i < 4; ++i) {
    int c = lane + 32 * i;
    float t = x[(size_t)row * 128 + c] + beta * rr[i] + (1.0f - beta) * o[i];
    hv[i] = t; s += t; s2 += t * t;
    h[(size_t)row * 128 + c] = t;
  }
  s = wave_sum32(s); s2 = wave_sum32(s2);
  float mu = s * (1.0f / 128.0f);
  float var = s2 * (1.0f / 128.0f) - mu * mu;
  float rs = rsqrtf(var + 1e-5f);
  #pragma unroll
  for (int i = 0; i < 4; ++i) {
    int c = lane + 32 * i;
    h2h[(size_t)row * 128 + c] = (h16)((hv[i] - mu) * rs * g2[c] + b2[c]);
  }
}

// --- FFN up-projection + exact GELU: [N,128] x [128,512] -> f16 -------------
__global__ void ffn1_k(const h16* __restrict__ h2h, const h16* __restrict__ W1t,
                       const float* __restrict__ bf1, h16* __restrict__ t, int nrows) {
  int wv = threadIdx.x >> 5, lane = threadIdx.x & 31;
  int row0 = blockIdx.x * 16;
  if (row0 >= nrows) return;
  int half = lane >> 4, mn = lane & 15;
  for (int tt = wv; tt < 32; tt += 8) {
    int col0 = tt * 16;
    v8f c = wmma_tile(h2h, 128, W1t, 128, row0, col0, 128, lane);
    int col = col0 + mn;
    float bb = bf1[col];
    #pragma unroll
    for (int r = 0; r < 8; ++r) {
      int m = row0 + half * 8 + r;
      float xg = c[r] + bb;
      float ge = 0.5f * xg * (1.0f + erff(xg * 0.70710678118654752f));
      t[(size_t)m * 512 + col] = (h16)ge;
    }
  }
}

// --- FFN down-projection + final residual: [N,512] x [512,128] --------------
__global__ void ffn2_k(const h16* __restrict__ t, const h16* __restrict__ W2t,
                       const float* __restrict__ bf2, const float* __restrict__ h,
                       float* __restrict__ out, int nrows) {
  int wv = threadIdx.x >> 5, lane = threadIdx.x & 31;
  int row0 = blockIdx.x * 16;
  if (row0 >= nrows) return;
  int half = lane >> 4, mn = lane & 15;
  int col0 = wv * 16;  // 8 waves cover 128 output cols
  v8f c = wmma_tile(t, 512, W2t, 512, row0, col0, 512, lane);
  int col = col0 + mn;
  float bb = bf2[col];
  #pragma unroll
  for (int r = 0; r < 8; ++r) {
    int m = row0 + half * 8 + r;
    out[(size_t)m * 128 + col] = h[(size_t)m * 128 + col] + c[r] + bb;
  }
}

// ---------------------------------------------------------------------------
extern "C" void kernel_launch(void* const* d_in, const int* in_sizes, int n_in,
                              void* d_out, int out_size, void* d_ws, size_t ws_size,
                              hipStream_t stream) {
  const float* x       = (const float*)d_in[0];
  const int*   ei      = (const int*)  d_in[1];
  const int*   et      = (const int*)  d_in[2];
  const float* rel_emb = (const float*)d_in[3];
  const float* Wq = (const float*)d_in[4];  const float* bq = (const float*)d_in[5];
  const float* Wk = (const float*)d_in[6];  const float* bk = (const float*)d_in[7];
  const float* Wv = (const float*)d_in[8];  const float* bv = (const float*)d_in[9];
  const float* We = (const float*)d_in[10];
  const float* Ws = (const float*)d_in[11]; const float* bs = (const float*)d_in[12];
  const float* wb = (const float*)d_in[13];
  const float* g1 = (const float*)d_in[14]; const float* b1 = (const float*)d_in[15];
  const float* g2 = (const float*)d_in[16]; const float* b2 = (const float*)d_in[17];
  const float* W1 = (const float*)d_in[18]; const float* bf1 = (const float*)d_in[19];
  const float* W2 = (const float*)d_in[20]; const float* bf2 = (const float*)d_in[21];
  float* out = (float*)d_out;

  // workspace layout (256B aligned)
  char* base = (char*)d_ws;
  size_t o = 0;
  auto alloc = [&](size_t bytes) { size_t r = o; o += (bytes + 255) & ~(size_t)255; return r; };
  size_t o_wqkvs = alloc(512 * 128 * sizeof(h16));
  size_t o_w1t   = alloc(512 * 128 * sizeof(h16));
  size_t o_w2t   = alloc(512 * 128 * sizeof(h16));
  size_t o_relp  = alloc(NREL * 128 * sizeof(float));
  size_t o_xnh   = alloc((size_t)NNODES * 128 * sizeof(h16));
  size_t o_q     = alloc((size_t)NNODES * 128 * sizeof(float));
  size_t o_k     = alloc((size_t)NNODES * 128 * sizeof(float));
  size_t o_v     = alloc((size_t)NNODES * 128 * sizeof(float));
  size_t o_xr    = alloc((size_t)NNODES * 128 * sizeof(float));
  size_t o_alpha = alloc((size_t)NEDGES * 4 * sizeof(float));
  size_t o_nmax  = alloc((size_t)NNODES * 4 * sizeof(float));
  size_t o_nsum  = alloc((size_t)NNODES * 4 * sizeof(float));
  size_t o_acc   = alloc((size_t)NNODES * 128 * sizeof(float));
  size_t o_h     = alloc((size_t)NNODES * 128 * sizeof(float));
  size_t o_h2h   = alloc((size_t)NNODES * 128 * sizeof(h16));
  size_t o_t     = o_q;  // alias: q & k (51.2 MB) are dead before ffn1 writes t

  h16*   Wqkvs_t = (h16*)(base + o_wqkvs);
  h16*   W1t     = (h16*)(base + o_w1t);
  h16*   W2t     = (h16*)(base + o_w2t);
  float* relp    = (float*)(base + o_relp);
  h16*   xnh     = (h16*)(base + o_xnh);
  float* q       = (float*)(base + o_q);
  float* k       = (float*)(base + o_k);
  float* v       = (float*)(base + o_v);
  float* xr      = (float*)(base + o_xr);
  float* alpha   = (float*)(base + o_alpha);
  float* nmax    = (float*)(base + o_nmax);
  float* nsum    = (float*)(base + o_nsum);
  float* acc     = (float*)(base + o_acc);
  float* h       = (float*)(base + o_h);
  h16*   h2h     = (h16*)(base + o_h2h);
  h16*   t       = (h16*)(base + o_t);

  // 1. weight packing (f32 -> f16 transposed-packed) + rel_proj table
  pack_weights_k<<<256, 256, 0, stream>>>(Wq, Wk, Wv, Ws, W1, W2, Wqkvs_t, W1t, W2t);
  relproj_k<<<1, 128, 0, stream>>>(rel_emb, We, relp);

  // 2. norm1 -> f16
  ln1_k<<<NNODES / 8, 256, 0, stream>>>(x, g1, b1, xnh, NNODES);

  // 3. fused Q|K|V|skip projection (WMMA)
  gemm_qkvs_k<<<NNODES / 16, 256, 0, stream>>>(xnh, Wqkvs_t, bq, bk, bv, bs,
                                               q, k, v, xr, NNODES);

  // 4. segment softmax + attention aggregation over edges
  init_k<<<(NNODES * 128 + 255) / 256, 256, 0, stream>>>(nmax, nsum, acc);
  edge_alpha_k<<<NEDGES / 8, 256, 0, stream>>>(ei, et, q, k, relp, alpha);
  segmax_k<<<(NEDGES * 4 + 255) / 256, 256, 0, stream>>>(ei, alpha, nmax);
  expsum_k<<<(NEDGES * 4 + 255) / 256, 256, 0, stream>>>(ei, alpha, nmax, nsum);
  agg_k<<<NEDGES / 8, 256, 0, stream>>>(ei, et, alpha, nsum, v, relp, acc);

  // 5. beta gate + residual + norm2
  gate_k<<<NNODES / 8, 256, 0, stream>>>(x, acc, xr, wb, g2, b2, h, h2h, NNODES);

  // 6. FFN (WMMA) + final residual
  ffn1_k<<<NNODES / 16, 256, 0, stream>>>(h2h, W1t, bf1, t, NNODES);
  ffn2_k<<<NNODES / 16, 256, 0, stream>>>(t, W2t, bf2, h, out, NNODES);

  (void)in_sizes; (void)n_in; (void)out_size; (void)ws_size;
}
